// multiHeadAttention_56916906607263
// MI455X (gfx1250) — matrix operand order repaired and checked by reading, not compiled, not run on hardware
//
#include <hip/hip_runtime.h>
#include <hip/hip_bf16.h>

#define EMBED 1024
#define HEADS 16
#define HDIM  64
#define SEQ   2048
#define BATCH 2

typedef __attribute__((ext_vector_type(16))) _Float16 v16h;
typedef __attribute__((ext_vector_type(8)))  _Float16 v8h;
typedef __attribute__((ext_vector_type(8)))  float    v8f;

#if __has_builtin(__builtin_amdgcn_tensor_load_to_lds) && __has_builtin(__builtin_amdgcn_s_wait_tensorcnt)
#define USE_TDM 1
#else
#define USE_TDM 0
#endif

// ---- fragment loaders matching CDNA5 wave32 WMMA VGPR layouts ----
// A (16x32 f16): lanes0-15 hold K=[0..7],[16..23] of row M=lane; lanes16-31 K=[8..15],[24..31].
// caller passes base = rowptr + k0 + 8*(lane>>4); halfs0-7 at base, halfs8-15 at base+16.
static __device__ __forceinline__ v16h ld_a_frag(const _Float16* base) {
  v8h lo = *(const v8h*)(base);
  v8h hi = *(const v8h*)(base + 16);
  return __builtin_shufflevector(lo, hi, 0,1,2,3,4,5,6,7,8,9,10,11,12,13,14,15);
}
// B (32x16 f16): column N=lane&15; lanes0-15 K=[0..7],[16..23]; lanes16-31 K=[8..15],[24..31] (base K 8*(lane>>4)).
static __device__ __forceinline__ v16h ld_b_frag(const _Float16* base) {
  v8h lo = *(const v8h*)(base);
  v8h hi = *(const v8h*)(base + 16);
  return __builtin_shufflevector(lo, hi, 0,1,2,3,4,5,6,7,8,9,10,11,12,13,14,15);
}
static __device__ __forceinline__ v8f wmma32(v16h a, v16h b, v8f c) {
  return __builtin_amdgcn_wmma_f32_16x16x32_f16(false, a, false, b, (short)0, c, false, false);
}

#if USE_TDM
typedef __attribute__((ext_vector_type(4))) unsigned int v4u;
typedef __attribute__((ext_vector_type(8))) int v8i;
typedef __attribute__((ext_vector_type(4))) int v4i;

static __device__ __forceinline__ int rfl(unsigned x) {
  return __builtin_amdgcn_readfirstlane((int)x);
}
// 2-D TDM tile load, descriptor packed per CDNA5 D# (groups 0/1; trailing groups zero => 2D).
// td0/td1 = tensor dims (data_size units), tile0/tile1 = tile dims, stride0 = row stride.
static __device__ __forceinline__ void tdm_load_2d(
    unsigned lds_byte_off, const void* gptr,
    unsigned td0, unsigned td1, unsigned tile0, unsigned tile1, unsigned stride0)
{
  unsigned long long ga = (unsigned long long)(size_t)gptr;
  v4u g0 = { (unsigned)rfl(1u),                               // count=1, user mode
             (unsigned)rfl(lds_byte_off),                     // lds_addr
             (unsigned)rfl((unsigned)ga),                     // global_addr[31:0]
             (unsigned)rfl((unsigned)(((ga >> 32) & 0x1FFFFFFull) | (2u << 30))) }; // [56:32] | type=2
  v8i g1 = { rfl(1u << 16),                                   // data_size=1 (2 bytes), mask=0
             rfl(td0 << 16),                                  // tensor_dim0[15:0] @bit48
             rfl(((td1 & 0xFFFFu) << 16) | (td0 >> 16)),      // tensor_dim1 lo | tensor_dim0 hi
             rfl((tile0 << 16) | (td1 >> 16)),                // tile_dim0 | tensor_dim1 hi
             rfl(tile1),                                      // tile_dim1 (tile_dim2=0)
             rfl(stride0),                                    // tensor_dim0_stride lo32
             rfl(0), rfl(0) };                                // stride hi / dim1_stride = 0
  v4i gz4 = { 0, 0, 0, 0 };
  v8i gz8 = { 0, 0, 0, 0, 0, 0, 0, 0 };
  __builtin_amdgcn_tensor_load_to_lds(g0, g1, gz4, gz4, gz8, 0);
}
static __device__ __forceinline__ unsigned lds_off_of(const void* p) {
  return (unsigned)(size_t)p;   // generic LDS pointer: low 32 bits = wave-relative LDS offset
}
#endif

// ---- f32 -> f16 conversion ----
__global__ __launch_bounds__(256) void cvt_f32_f16(const float* __restrict__ in,
                                                   _Float16* __restrict__ out, int n) {
  int i = (blockIdx.x * blockDim.x + threadIdx.x) * 4;
  if (i + 3 < n) {
    float4 f = *(const float4*)(in + i);
    out[i + 0] = (_Float16)f.x;
    out[i + 1] = (_Float16)f.y;
    out[i + 2] = (_Float16)f.z;
    out[i + 3] = (_Float16)f.w;
  }
}

// ---- C = A[M,K] * W[N,K]^T + bias, WMMA f16->f32 ----
// mode 0: f32 out[m*N+n]
// mode 1: f16 out[((b*H+h)*SEQ+s)*HDIM+d]   (Q,K layout [B,H,S,D])
// mode 2: f16 out[((b*H+h)*HDIM+d)*SEQ+s]   (V layout  [B,H,D,S])
__global__ __launch_bounds__(256) void gemm_nt_wmma(
    const _Float16* __restrict__ A, const _Float16* __restrict__ W,
    const float* __restrict__ bias, void* __restrict__ out,
    int M, int N, int K, int nb128, int mode)
{
  const int wave  = threadIdx.x >> 5;
  const int lane  = threadIdx.x & 31;
  const int hlf   = lane >> 4;
  const int l16   = lane & 15;
  const int bm    = blockIdx.x / nb128;
  const int bn    = blockIdx.x % nb128;
  const int waveM = wave >> 1;
  const int waveN = wave & 1;
  const int rowA0 = bm * 128 + waveM * 32;
  const int colB0 = bn * 128 + waveN * 64;

  const v8f zacc = {0.f,0.f,0.f,0.f,0.f,0.f,0.f,0.f};
  v8f acc[2][4];
  #pragma unroll
  for (int mi = 0; mi < 2; ++mi)
    #pragma unroll
    for (int ni = 0; ni < 4; ++ni) acc[mi][ni] = zacc;

  const _Float16* Abase = A + (size_t)(rowA0 + l16) * K + 8 * hlf;
  const _Float16* Wbase = W + (size_t)(colB0 + l16) * K + 8 * hlf;

  for (int k0 = 0; k0 < K; k0 += 32) {
    if (k0 + 32 < K) {  // lowers to global_prefetch_b8
      __builtin_prefetch(Abase + k0 + 32, 0, 3);
      __builtin_prefetch(Abase + (size_t)16 * K + k0 + 32, 0, 3);
      __builtin_prefetch(Wbase + k0 + 32, 0, 3);
      __builtin_prefetch(Wbase + (size_t)48 * K + k0 + 32, 0, 3);
    }
    v16h a0 = ld_a_frag(Abase + k0);
    v16h a1 = ld_a_frag(Abase + (size_t)16 * K + k0);
    v16h b[4];
    #pragma unroll
    for (int ni = 0; ni < 4; ++ni)
      b[ni] = ld_b_frag(Wbase + (size_t)ni * 16 * K + k0);
    #pragma unroll
    for (int ni = 0; ni < 4; ++ni) {
      acc[0][ni] = wmma32(a0, b[ni], acc[0][ni]);
      acc[1][ni] = wmma32(a1, b[ni], acc[1][ni]);
    }
  }

  #pragma unroll
  for (int mi = 0; mi < 2; ++mi)
    #pragma unroll
    for (int ni = 0; ni < 4; ++ni) {
      const int n = colB0 + ni * 16 + l16;
      const float bb = bias[n];
      #pragma unroll
      for (int r = 0; r < 8; ++r) {
        const int m = rowA0 + mi * 16 + r + 8 * hlf;
        const float v = acc[mi][ni][r] + bb;
        if (mode == 0) {
          ((float*)out)[(size_t)m * N + n] = v;
        } else {
          const int bI = m / SEQ, s = m % SEQ;
          const int h  = n / HDIM, d = n % HDIM;
          const _Float16 hv = (_Float16)v;
          if (mode == 1)
            ((_Float16*)out)[((size_t)(bI * HEADS + h) * SEQ + s) * HDIM + d] = hv;
          else
            ((_Float16*)out)[((size_t)(bI * HEADS + h) * HDIM + d) * SEQ + s] = hv;
        }
      }
    }
}

// ---- flash-style attention ----
// Block = 8 waves sharing one (b,h); wave w handles q-tile qt0+w (16 rows).
// K/V kv-chunks (32 wide) are staged in LDS via the Tensor Data Mover, double-buffered.
__global__ __launch_bounds__(256) void attn_wmma(
    const _Float16* __restrict__ Q,   // [B,H,S,D]
    const _Float16* __restrict__ Kt,  // [B,H,S,D]
    const _Float16* __restrict__ Vt,  // [B,H,D,S]
    const int* __restrict__ mask,     // [B,1,1,S]
    _Float16* __restrict__ Obuf)      // [B,S,EMBED] f16
{
  __shared__ __align__(16) _Float16 plds[8][16 * 32];
#if USE_TDM
  __shared__ __align__(128) _Float16 stK[2][32 * HDIM]; // [kv row][d]
  __shared__ __align__(128) _Float16 stV[2][HDIM * 32]; // [d][kv col]
#endif
  const int wave = threadIdx.x >> 5;
  const int lane = threadIdx.x & 31;
  const int hlf  = lane >> 4;
  const int l16  = lane & 15;
  const int bh = blockIdx.x >> 4;            // b*HEADS + h
  const int b  = bh >> 4;
  const int h  = bh & 15;
  const int qt = (blockIdx.x & 15) * 8 + wave;

  const _Float16* Qb = Q  + ((size_t)bh * SEQ + qt * 16) * HDIM;
  const _Float16* Kb = Kt + (size_t)bh * SEQ * HDIM;
  const _Float16* Vb = Vt + (size_t)bh * HDIM * SEQ;
  const int* mb = mask + b * SEQ;
  _Float16* pw = plds[wave];

  const v16h aq0 = ld_a_frag(Qb + (size_t)l16 * HDIM + 0  + 8 * hlf);
  const v16h aq1 = ld_a_frag(Qb + (size_t)l16 * HDIM + 32 + 8 * hlf);

  const v8f zacc = {0.f,0.f,0.f,0.f,0.f,0.f,0.f,0.f};
  v8f o[4] = {zacc, zacc, zacc, zacc};
  float mrow[8], lrow[8];
  #pragma unroll
  for (int r = 0; r < 8; ++r) { mrow[r] = -3.0e38f; lrow[r] = 0.f; }

  const float scale = 0.03125f; // 1/sqrt(EMBED)
  const int NIT = SEQ / 32;

#if USE_TDM
  if (wave == 0) {   // preload chunk 0 into buffer 0
    tdm_load_2d(lds_off_of(&stK[0][0]), Kb,      HDIM, 32, HDIM, 32, HDIM);
    tdm_load_2d(lds_off_of(&stV[0][0]), Vb,      32, HDIM, 32, HDIM, SEQ);
  }
#endif

  for (int it = 0; it < NIT; ++it) {
    const int j0 = it * 32;
    v8f sc0 = zacc, sc1 = zacc;
#if USE_TDM
    const int cur = it & 1;
    __syncthreads();                       // all reads of buf[cur^1] finished
    if (wave == 0) {
      if (it + 1 < NIT) {
        const _Float16* kg = Kb + (size_t)(j0 + 32) * HDIM;
        const _Float16* vg = Vb + (j0 + 32);
        tdm_load_2d(lds_off_of(&stK[cur ^ 1][0]), kg, HDIM, 32, HDIM, 32, HDIM);
        tdm_load_2d(lds_off_of(&stV[cur ^ 1][0]), vg, 32, HDIM, 32, HDIM, SEQ);
        __builtin_amdgcn_s_wait_tensorcnt(2);  // current buffer's 2 TDM ops done
      } else {
        __builtin_amdgcn_s_wait_tensorcnt(0);
      }
    }
    __syncthreads();                       // buf[cur] published to all waves
    asm volatile("" ::: "memory");
    size_t skp = (size_t)&stK[cur][0];
    size_t svp = (size_t)&stV[cur][0];
    asm volatile("" : "+s"(skp));          // hide provenance: TDM wrote this LDS
    asm volatile("" : "+s"(svp));
    const _Float16* sk = (const _Float16*)skp;
    const _Float16* sv = (const _Float16*)svp;
    {
      v16h b00 = ld_b_frag(sk + (size_t)l16 * HDIM + 8 * hlf);
      v16h b01 = ld_b_frag(sk + (size_t)l16 * HDIM + 32 + 8 * hlf);
      v16h b10 = ld_b_frag(sk + (size_t)(16 + l16) * HDIM + 8 * hlf);
      v16h b11 = ld_b_frag(sk + (size_t)(16 + l16) * HDIM + 32 + 8 * hlf);
      sc0 = wmma32(aq0, b00, sc0); sc0 = wmma32(aq1, b01, sc0);
      sc1 = wmma32(aq0, b10, sc1); sc1 = wmma32(aq1, b11, sc1);
    }
#else
    {
      const _Float16* k0p = Kb + (size_t)(j0 + l16) * HDIM + 8 * hlf;
      const _Float16* k1p = Kb + (size_t)(j0 + 16 + l16) * HDIM + 8 * hlf;
      v16h b00 = ld_b_frag(k0p);
      v16h b01 = ld_b_frag(k0p + 32);
      v16h b10 = ld_b_frag(k1p);
      v16h b11 = ld_b_frag(k1p + 32);
      sc0 = wmma32(aq0, b00, sc0); sc0 = wmma32(aq1, b01, sc0);
      sc1 = wmma32(aq0, b10, sc1); sc1 = wmma32(aq1, b11, sc1);
    }
#endif
    const int mk0 = mb[j0 + l16];
    const int mk1 = mb[j0 + 16 + l16];

    float p0[8], p1[8];
    #pragma unroll
    for (int r = 0; r < 8; ++r) {
      float x0 = mk0 ? sc0[r] * scale : -1e20f * scale;
      float x1 = mk1 ? sc1[r] * scale : -1e20f * scale;
      p0[r] = x0; p1[r] = x1;
      float cm = fmaxf(x0, x1);                    // row reduce across 16-lane half
      cm = fmaxf(cm, __shfl_xor(cm, 1));
      cm = fmaxf(cm, __shfl_xor(cm, 2));
      cm = fmaxf(cm, __shfl_xor(cm, 4));
      cm = fmaxf(cm, __shfl_xor(cm, 8));
      const float mnew  = fmaxf(mrow[r], cm);
      const float alpha = __expf(mrow[r] - mnew);
      mrow[r] = mnew;
      const float e0 = __expf(p0[r] - mnew);
      const float e1 = __expf(p1[r] - mnew);
      p0[r] = e0; p1[r] = e1;
      float rs = e0 + e1;
      rs += __shfl_xor(rs, 1);
      rs += __shfl_xor(rs, 2);
      rs += __shfl_xor(rs, 4);
      rs += __shfl_xor(rs, 8);
      lrow[r] = lrow[r] * alpha + rs;
      #pragma unroll
      for (int td = 0; td < 4; ++td) o[td][r] *= alpha;
    }

    // C-layout -> A-fragment relayout of P through per-wave private LDS
    #pragma unroll
    for (int r = 0; r < 8; ++r) {
      const int Mr = r + 8 * hlf;
      pw[Mr * 32 + l16]      = (_Float16)p0[r];
      pw[Mr * 32 + 16 + l16] = (_Float16)p1[r];
    }
    asm volatile("s_wait_dscnt 0x0" ::: "memory");
    const v16h pa = ld_a_frag(pw + l16 * 32 + 8 * hlf);

    #pragma unroll
    for (int td = 0; td < 4; ++td) {
#if USE_TDM
      v16h bv = ld_b_frag(sv + (size_t)(td * 16 + l16) * 32 + 8 * hlf);
#else
      v16h bv = ld_b_frag(Vb + (size_t)(td * 16 + l16) * SEQ + j0 + 8 * hlf);
#endif
      o[td] = wmma32(pa, bv, o[td]);
    }
    asm volatile("" ::: "memory"); // keep next-iter LDS stores after these reads
  }

  #pragma unroll
  for (int td = 0; td < 4; ++td) {
    #pragma unroll
    for (int r = 0; r < 8; ++r) {
      const int srow = qt * 16 + r + 8 * hlf;
      const int col  = h * HDIM + td * 16 + l16;
      Obuf[((size_t)(b * SEQ + srow)) * EMBED + col] = (_Float16)(o[td][r] / lrow[r]);
    }
  }
}

extern "C" void kernel_launch(void* const* d_in, const int* in_sizes, int n_in,
                              void* d_out, int out_size, void* d_ws, size_t ws_size,
                              hipStream_t stream) {
  const float* query = (const float*)d_in[0];
  const float* key_  = (const float*)d_in[1];
  const float* value = (const float*)d_in[2];
  const int*   mask  = (const int*)d_in[3];
  const float* Wq = (const float*)d_in[4];
  const float* bq = (const float*)d_in[5];
  const float* Wk = (const float*)d_in[6];
  const float* bk = (const float*)d_in[7];
  const float* Wv = (const float*)d_in[8];
  const float* bv = (const float*)d_in[9];
  const float* Wo = (const float*)d_in[10];
  const float* bo = (const float*)d_in[11];

  const size_t ACT = (size_t)BATCH * SEQ * EMBED; // 4,194,304 elements
  const size_t WEL = (size_t)EMBED * EMBED;       // 1,048,576 elements
  char* ws = (char*)d_ws;
  size_t off = 0;
  auto carve = [&](size_t bytes) { void* p = ws + off; off += bytes; return p; };
  _Float16* xq  = (_Float16*)carve(ACT * 2);
  _Float16* xk  = (_Float16*)carve(ACT * 2);
  _Float16* xv  = (_Float16*)carve(ACT * 2);
  _Float16* wqh = (_Float16*)carve(WEL * 2);
  _Float16* wkh = (_Float16*)carve(WEL * 2);
  _Float16* wvh = (_Float16*)carve(WEL * 2);
  _Float16* woh = (_Float16*)carve(WEL * 2);
  _Float16* Qh  = (_Float16*)carve(ACT * 2);
  _Float16* Kh  = (_Float16*)carve(ACT * 2);
  _Float16* Vth = (_Float16*)carve(ACT * 2);
  _Float16* Ah  = (_Float16*)carve(ACT * 2);

  // stage 1: f32 -> f16 conversions
  cvt_f32_f16<<<(int)(ACT / 4 / 256), 256, 0, stream>>>(query, xq, (int)ACT);
  cvt_f32_f16<<<(int)(ACT / 4 / 256), 256, 0, stream>>>(key_,  xk, (int)ACT);
  cvt_f32_f16<<<(int)(ACT / 4 / 256), 256, 0, stream>>>(value, xv, (int)ACT);
  cvt_f32_f16<<<(int)(WEL / 4 / 256), 256, 0, stream>>>(Wq, wqh, (int)WEL);
  cvt_f32_f16<<<(int)(WEL / 4 / 256), 256, 0, stream>>>(Wk, wkh, (int)WEL);
  cvt_f32_f16<<<(int)(WEL / 4 / 256), 256, 0, stream>>>(Wv, wvh, (int)WEL);
  cvt_f32_f16<<<(int)(WEL / 4 / 256), 256, 0, stream>>>(Wo, woh, (int)WEL);

  // stage 2: projections (M=4096, N=1024, K=1024), 128x128 block tiles
  const int M = BATCH * SEQ, N = EMBED, K = EMBED;
  const int nb128 = N / 128;
  const int gblocks = (M / 128) * nb128; // 256
  gemm_nt_wmma<<<gblocks, 256, 0, stream>>>(xq, wqh, bq, Qh,  M, N, K, nb128, 1);
  gemm_nt_wmma<<<gblocks, 256, 0, stream>>>(xk, wkh, bk, Kh,  M, N, K, nb128, 1);
  gemm_nt_wmma<<<gblocks, 256, 0, stream>>>(xv, wvh, bv, Vth, M, N, K, nb128, 2);

  // stage 3: attention (each block = one (b,h) slice x 8 q-tiles)
  attn_wmma<<<512, 256, 0, stream>>>(Qh, Kh, Vth, mask, Ah);

  // stage 4: output projection, f32 result + bias
  gemm_nt_wmma<<<gblocks, 256, 0, stream>>>(Ah, woh, bo, d_out, M, N, K, nb128, 0);
}
